// DeepseekMoE_21904333210008
// MI455X (gfx1250) — matrix operand order repaired and checked by reading, not compiled, not run on hardware
//
#include <hip/hip_runtime.h>

// ---------------- problem constants ----------------
#define E_  32      // experts
#define K_  6       // top-k
#define H_  2048    // hidden
#define I_  1408    // routed intermediate (also the virtual-chunk size of shared)
#define SI_ 2816    // shared expert intermediate
#define T_  1024    // tokens
#define MT_ 32      // tokens per block tile

typedef __attribute__((ext_vector_type(16))) __bf16 v16bf;
typedef __attribute__((ext_vector_type(8)))  __bf16 v8bf;
typedef __attribute__((ext_vector_type(8)))  float  v8f;

// fast sigmoid: one v_exp_f32 + one v_rcp_f32
__device__ __forceinline__ float fast_sigmoid(float g) {
  return __builtin_amdgcn_rcpf(1.0f + __expf(-g));
}

// load 16 contiguous fp32 -> bf16 fragment (B operand, K-contiguous per lane)
__device__ __forceinline__ v16bf load_cvt16(const float* __restrict__ p) {
  const float4* q = (const float4*)p;
  float4 f0 = q[0], f1 = q[1], f2 = q[2], f3 = q[3];
  v16bf r;
  r[0]  = (__bf16)f0.x; r[1]  = (__bf16)f0.y; r[2]  = (__bf16)f0.z; r[3]  = (__bf16)f0.w;
  r[4]  = (__bf16)f1.x; r[5]  = (__bf16)f1.y; r[6]  = (__bf16)f1.z; r[7]  = (__bf16)f1.w;
  r[8]  = (__bf16)f2.x; r[9]  = (__bf16)f2.y; r[10] = (__bf16)f2.z; r[11] = (__bf16)f2.w;
  r[12] = (__bf16)f3.x; r[13] = (__bf16)f3.y; r[14] = (__bf16)f3.z; r[15] = (__bf16)f3.w;
  return r;
}

// ---------------- workspace / output init ----------------
__global__ void moe_init_kernel(int* __restrict__ cnt,
                                int* __restrict__ tok_list,
                                float* __restrict__ wgt_list,
                                float* __restrict__ out) {
  int i = blockIdx.x * blockDim.x + threadIdx.x;
  if (i < T_ * H_) out[i] = 0.0f;
  if (i < E_ * T_) { tok_list[i] = 0; wgt_list[i] = 0.0f; }
  if (i < E_) cnt[i] = 0;
}

// ---------------- gating: softmax + top-k + compaction ----------------
// one wave per token; lane == expert id (E == wave32)
__global__ __launch_bounds__(256) void moe_gate_kernel(
    const float* __restrict__ x, const float* __restrict__ gw,
    int* __restrict__ cnt, int* __restrict__ tok_list, float* __restrict__ wgt_list) {
  const int lane = threadIdx.x & 31;     // expert
  const int wv   = threadIdx.x >> 5;
  const int t    = blockIdx.x * 8 + wv;  // token
  if (t >= T_) return;

  const float* xr = x + (size_t)t * H_;
  const float* wr = gw + (size_t)lane * H_;
  float acc = 0.0f;
  #pragma unroll 4
  for (int h = 0; h < H_; h += 4) {
    float4 xv = *(const float4*)(xr + h);
    float4 wv4 = *(const float4*)(wr + h);
    acc += xv.x * wv4.x + xv.y * wv4.y + xv.z * wv4.z + xv.w * wv4.w;
  }
  // wave softmax over 32 experts
  float m = acc;
  for (int off = 16; off; off >>= 1) m = fmaxf(m, __shfl_xor(m, off, 32));
  float e = __expf(acc - m);
  float s = e;
  for (int off = 16; off; off >>= 1) s += __shfl_xor(s, off, 32);
  const float p = e * __builtin_amdgcn_rcpf(s);

  // iterative top-K via ballot (NORM_TOPK_PROB == false -> weight = p)
  float pw = p;
  for (int it = 0; it < K_; ++it) {
    float mx = pw;
    for (int off = 16; off; off >>= 1) mx = fmaxf(mx, __shfl_xor(mx, off, 32));
    unsigned long long ball = __ballot(pw == mx);
    int sel = __ffsll(ball) - 1;
    if (lane == sel) {
      int idx = atomicAdd(&cnt[lane], 1);
      tok_list[lane * T_ + idx] = t;
      wgt_list[lane * T_ + idx] = p;
      pw = -1.0f;  // exclude from next rounds (probs are > 0)
    }
  }
}

// ---------------- fused SwiGLU expert MLP (M = 32 tokens/block) ----------------
// grid: (token_tile, expert_or_chunk). block: 256 threads = 8 waves.
// mode 0: routed expert blockIdx.y, gathered token list
// mode 1: shared expert virtual chunk blockIdx.y (I-dim split in halves of 1408)
// All results atomically accumulate into zero-initialized out.
__global__ __launch_bounds__(256) void moe_mlp_kernel(
    const float* __restrict__ x,
    const float* __restrict__ w1base,
    const float* __restrict__ w2base,
    const int* __restrict__ cnt,
    const int* __restrict__ tok_list,
    const float* __restrict__ wgt_list,
    float* __restrict__ out,
    int mode) {
  extern __shared__ char smem[];
  __bf16* xt  = (__bf16*)smem;                                // [MT][H] bf16
  __bf16* ht  = (__bf16*)(smem + (size_t)MT_ * H_ * 2);       // [MT][I] bf16
  float*  wgt = (float*)(smem + (size_t)MT_ * H_ * 2 + (size_t)MT_ * I_ * 2);  // [MT]
  int*    tok = (int*)(wgt + MT_);                            // [MT]

  const int tid  = threadIdx.x;
  const int lane = tid & 31;
  const int wv   = tid >> 5;
  const int tile = blockIdx.x;
  const int ex   = blockIdx.y;   // expert (mode 0) or shared chunk (mode 1)

  const float* w1;   // g-rows base; u-row at relative row offset uOff
  const float* w2;   // column-origin of this chunk; row stride w2s
  int uOff, w2s;
  if (mode == 0) {
    w1 = w1base + (size_t)ex * 2 * I_ * H_;
    w2 = w2base + (size_t)ex * H_ * I_;
    uOff = I_;  w2s = I_;
    if (tile * MT_ >= cnt[ex]) return;  // uniform early-out for empty tiles
    if (tid < MT_) {
      tok[tid] = tok_list[ex * T_ + tile * MT_ + tid];  // padded slots: tok=0, wgt=0
      wgt[tid] = wgt_list[ex * T_ + tile * MT_ + tid];
    }
  } else {
    w1 = w1base + (size_t)ex * I_ * H_;   // chunk's g rows
    w2 = w2base + (size_t)ex * I_;        // chunk's w2 columns
    uOff = SI_; w2s = SI_;
    if (tid < MT_) { tok[tid] = tile * MT_ + tid; wgt[tid] = 1.0f; }
  }
  __syncthreads();

  // stage the 32-token activation tile in LDS as bf16:
  // 8 threads per row, float4 global loads, contiguous bf16 LDS stores
  {
    const int m   = tid >> 3;
    const int sub = tid & 7;
    const float* src = x + (size_t)tok[m] * H_;
    __bf16* dst = xt + (size_t)m * H_;
    #pragma unroll 4
    for (int c = sub * 4; c < H_; c += 32) {
      float4 f = *(const float4*)(src + c);
      dst[c + 0] = (__bf16)f.x; dst[c + 1] = (__bf16)f.y;
      dst[c + 2] = (__bf16)f.z; dst[c + 3] = (__bf16)f.w;
    }
  }
  __syncthreads();

  const int mlo = lane & 15;    // A row / B,D col
  const int hi  = lane >> 4;    // half-wave select

  // ---- phase 1: gu = X @ w1^T, h = silu(g)*u  (stage h in LDS as bf16) ----
  for (int jt = wv; jt < (I_ >> 4); jt += 8) {
    v8f accg0 = {}, accu0 = {}, accg1 = {}, accu1 = {};
    const int ng = jt * 16 + mlo;                 // gate-row of w1 for this lane
    const float* bg = w1 + (size_t)ng * H_;
    const float* bu = w1 + (size_t)(ng + uOff) * H_;
    for (int k = 0; k < H_; k += 32) {
      const int kb = k + hi * 16;                 // B: 16 contiguous K elems
      __builtin_prefetch(bg + kb + 512, 0, 0);    // ~2KB ahead on each stream
      __builtin_prefetch(bu + kb + 512, 0, 0);
      const int ka = k + hi * 16;                 // A: 16 contiguous K elems
      union { v16bf v; v8bf h2[2]; } a0, a1;
      a0.h2[0] = *(const v8bf*)(xt + (size_t)mlo * H_ + ka);
      a0.h2[1] = *(const v8bf*)(xt + (size_t)mlo * H_ + ka + 8);
      a1.h2[0] = *(const v8bf*)(xt + (size_t)(mlo + 16) * H_ + ka);
      a1.h2[1] = *(const v8bf*)(xt + (size_t)(mlo + 16) * H_ + ka + 8);
      v16bf bgv = load_cvt16(bg + kb);
      v16bf buv = load_cvt16(bu + kb);
      accg0 = __builtin_amdgcn_wmma_f32_16x16x32_bf16(false, a0.v, false, bgv,
                                                      (short)0, accg0, false, false);
      accu0 = __builtin_amdgcn_wmma_f32_16x16x32_bf16(false, a0.v, false, buv,
                                                      (short)0, accu0, false, false);
      accg1 = __builtin_amdgcn_wmma_f32_16x16x32_bf16(false, a1.v, false, bgv,
                                                      (short)0, accg1, false, false);
      accu1 = __builtin_amdgcn_wmma_f32_16x16x32_bf16(false, a1.v, false, buv,
                                                      (short)0, accu1, false, false);
    }
    #pragma unroll
    for (int r = 0; r < 8; ++r) {
      const int m = r + hi * 8;                   // D row within 16-row half
      float g0 = accg0[r], u0 = accu0[r];
      float g1 = accg1[r], u1 = accu1[r];
      ht[(size_t)m * I_ + jt * 16 + mlo]        = (__bf16)(g0 * fast_sigmoid(g0) * u0);
      ht[(size_t)(m + 16) * I_ + jt * 16 + mlo] = (__bf16)(g1 * fast_sigmoid(g1) * u1);
    }
  }
  __syncthreads();

  // ---- phase 2: out_tile = h @ w2^T, scaled by gate weight, combined ----
  for (int nt = wv; nt < (H_ >> 4); nt += 8) {
    v8f acc0 = {}, acc1 = {};
    const int n = nt * 16 + mlo;
    const float* br = w2 + (size_t)n * w2s;
    for (int k = 0; k < I_; k += 32) {
      const int kb = k + hi * 16;
      __builtin_prefetch(br + kb + 512, 0, 0);
      const int ka = k + hi * 16;
      union { v16bf v; v8bf h2[2]; } a0, a1;
      a0.h2[0] = *(const v8bf*)(ht + (size_t)mlo * I_ + ka);
      a0.h2[1] = *(const v8bf*)(ht + (size_t)mlo * I_ + ka + 8);
      a1.h2[0] = *(const v8bf*)(ht + (size_t)(mlo + 16) * I_ + ka);
      a1.h2[1] = *(const v8bf*)(ht + (size_t)(mlo + 16) * I_ + ka + 8);
      v16bf bv = load_cvt16(br + kb);
      acc0 = __builtin_amdgcn_wmma_f32_16x16x32_bf16(false, a0.v, false, bv,
                                                     (short)0, acc0, false, false);
      acc1 = __builtin_amdgcn_wmma_f32_16x16x32_bf16(false, a1.v, false, bv,
                                                     (short)0, acc1, false, false);
    }
    #pragma unroll
    for (int r = 0; r < 8; ++r) {
      const int m = r + hi * 8;
      atomicAdd(out + (size_t)tok[m] * H_ + n,      acc0[r] * wgt[m]);
      atomicAdd(out + (size_t)tok[m + 16] * H_ + n, acc1[r] * wgt[m + 16]);
    }
  }
}

// ---------------- host launch ----------------
extern "C" void kernel_launch(void* const* d_in, const int* in_sizes, int n_in,
                              void* d_out, int out_size, void* d_ws, size_t ws_size,
                              hipStream_t stream) {
  const float* x   = (const float*)d_in[0];  // [T, H]
  const float* gw  = (const float*)d_in[1];  // [E, H]
  const float* w1  = (const float*)d_in[2];  // [E, 2I, H]
  const float* w2  = (const float*)d_in[3];  // [E, H, I]
  const float* sw1 = (const float*)d_in[4];  // [2*SI, H]
  const float* sw2 = (const float*)d_in[5];  // [H, SI]
  float* out = (float*)d_out;                // [T, H]

  int*   cnt      = (int*)d_ws;              // [E]
  int*   tok_list = cnt + E_;                // [E*T]
  float* wgt_list = (float*)(tok_list + E_ * T_);  // [E*T]

  // 1) zero output + compaction workspace
  moe_init_kernel<<<(T_ * H_ + 255) / 256, 256, 0, stream>>>(cnt, tok_list, wgt_list, out);

  // 2) gating: softmax + top-k + per-expert token compaction
  moe_gate_kernel<<<T_ / 8, 256, 0, stream>>>(x, gw, cnt, tok_list, wgt_list);

  // LDS: xt(32xH bf16) + ht(32xI bf16) + wgt/tok
  size_t smem = (size_t)MT_ * H_ * 2 + (size_t)MT_ * I_ * 2 + MT_ * 4 + MT_ * 4;

  // 3) shared expert as 2 virtual I=1408 chunks (atomic accumulate)
  moe_mlp_kernel<<<dim3(T_ / MT_, 2), 256, smem, stream>>>(
      x, sw1, sw2, nullptr, nullptr, nullptr, out, 1);

  // 4) routed experts (atomic accumulate)
  moe_mlp_kernel<<<dim3(T_ / MT_, E_), 256, smem, stream>>>(
      x, w1, w2, cnt, tok_list, wgt_list, out, 0);
}
